// TopKMoEAttention_15204184227984
// MI455X (gfx1250) — matrix operand order repaired and checked by reading, not compiled, not run on hardware
//
#include <hip/hip_runtime.h>

// ---------------- problem constants (match reference) ----------------
#define B_    4
#define T_    4096
#define D_    1024
#define H_    16
#define E_    8
#define SPAN_ 64
#define GH_   128
#define HD_   (D_ / H_)      // 64
#define NSPAN_ (T_ / SPAN_)  // 64 spans per batch
#define KSTEPS (D_ / 32)     // 32 K-blocks of 32

typedef __attribute__((ext_vector_type(16))) __bf16 v16bf;
typedef __attribute__((ext_vector_type(8)))  float  v8f;

// LDS strides (bf16 elements), chosen so every fragment load is 16B aligned
// and rows land on different banks.
#define XBS 40   // x block row stride (32 cols + pad)
#define WCS 40   // combined-weight block row stride
#define QS  72   // Q/K/Vt/P row stride (64 cols + pad)

struct SharedBuf {
  alignas(16) __bf16 xblk[SPAN_ * XBS];      //  5.0 KB  x block (64x32)
  alignas(16) __bf16 wc[3][HD_ * WCS];       // 15.0 KB  combined Wq/Wk/Wv blocks (64x32)
  alignas(16) __bf16 qb[SPAN_ * QS];         //  9.0 KB  Q (64x64)
  alignas(16) __bf16 kb[SPAN_ * QS];         //  9.0 KB  K (64x64)
  alignas(16) __bf16 vtb[HD_ * QS];          //  9.0 KB  V transposed (d-major, 64x64)
  alignas(16) __bf16 pb[4 * 16 * QS];        //  9.0 KB  P (probs) per wave
  alignas(16) float  maskf[SPAN_];           //  key mask for this span
};

// A-matrix fragment, 16x32 bf16 (ISA 7.12.2): lanes 0-15 hold rows 0-15 with
// K in {0..7, 16..23}; lanes 16-31 hold the same rows with K in {8..15, 24..31}.
__device__ inline v16bf load_frag_a(const __bf16* base, int stride, int lane) {
  const int row = lane & 15;
  const int kh  = (lane >> 4) << 3;          // 0 or 8
  const __bf16* p = base + row * stride + kh;
  union { v16bf v; float4 f[2]; } u;
  u.f[0] = *(const float4*)(p);              // K kh..kh+7
  u.f[1] = *(const float4*)(p + 16);         // K kh+16..kh+23
  return u.v;
}

// B-matrix fragment, 32x16 bf16: lane L supplies column (L&15); lanes 0-15
// hold K {0..7, 16..23}, lanes 16-31 hold K {8..15, 24..31} (matches A).
// Source layout: src[n][k] row-major with `stride`.
__device__ inline v16bf load_frag_b(const __bf16* base, int stride, int lane) {
  const int n  = lane & 15;
  const int kh = (lane >> 4) << 3;           // 0 or 8
  const __bf16* p = base + n * stride + kh;
  union { v16bf v; float4 f[2]; } u;
  u.f[0] = *(const float4*)(p);              // K kh..kh+7
  u.f[1] = *(const float4*)(p + 16);         // K kh+16..kh+23
  return u.v;
}

__device__ inline v8f wmma_bf16(v16bf a, v16bf b, v8f c) {
  // (neg_a, A, neg_b, B, c_mod, C, reuse_a, reuse_b)
  return __builtin_amdgcn_wmma_f32_16x16x32_bf16(false, a, false, b, (short)0, c,
                                                 false, false);
}

// ===================== kernel 1: gating (top-2 softmax) =====================
__global__ __launch_bounds__(128) void gate_kernel(
    const float* __restrict__ x, const int* __restrict__ am,
    const float* __restrict__ w1, const float* __restrict__ b1,
    const float* __restrict__ w2, const float* __restrict__ b2,
    float* __restrict__ gate_out /* per span: {e0,e1 (int bits), w0, w1} */) {
  __shared__ float pooled[D_];
  __shared__ float hbuf[GH_];
  __shared__ float logit[E_];
  __shared__ float msh[SPAN_];

  const int span = blockIdx.x;                 // 0..255
  const int b = span / NSPAN_, s = span % NSPAN_;
  const int tid = threadIdx.x;
  const float* xs = x + ((size_t)b * T_ + (size_t)s * SPAN_) * (size_t)D_;

  if (tid < SPAN_) msh[tid] = (float)am[b * T_ + s * SPAN_ + tid];
  __syncthreads();

  float msum = 0.f;
  for (int t = 0; t < SPAN_; ++t) msum += msh[t];
  const float minv = 1.0f / fmaxf(msum, 1e-9f);

  for (int c = tid; c < D_; c += 128) {
    float acc = 0.f;
    for (int t = 0; t < SPAN_; ++t) acc += xs[(size_t)t * D_ + c] * msh[t];
    pooled[c] = acc * minv;
  }
  __syncthreads();

  {  // h = relu(pooled @ w1^T + b1), one hidden unit per thread
    float acc = b1[tid];
    const float* wr = w1 + (size_t)tid * D_;
    for (int c = 0; c < D_; ++c) acc += pooled[c] * wr[c];
    hbuf[tid] = fmaxf(acc, 0.f);
  }
  __syncthreads();

  if (tid < E_) {
    float acc = b2[tid];
    const float* wr = w2 + tid * GH_;
    for (int j = 0; j < GH_; ++j) acc += hbuf[j] * wr[j];
    logit[tid] = acc;
  }
  __syncthreads();

  if (tid == 0) {
    int i0 = 0; float v0 = logit[0];
    for (int e = 1; e < E_; ++e) if (logit[e] > v0) { v0 = logit[e]; i0 = e; }
    int i1 = (i0 == 0) ? 1 : 0; float v1 = logit[i1];
    for (int e = 0; e < E_; ++e)
      if (e != i0 && logit[e] > v1) { v1 = logit[e]; i1 = e; }
    const float e1 = __expf(v1 - v0);
    const float w0 = 1.0f / (1.0f + e1);
    float* o = gate_out + span * 4;
    ((int*)o)[0] = i0;
    ((int*)o)[1] = i1;
    o[2] = w0;
    o[3] = e1 * w0;
  }
}

// ============ kernel 2: fused top-2 MoE projection + span attention ============
// One block per (span, head). 128 threads = 4 wave32; wave w owns query rows
// [16w, 16w+16). All matmuls via v_wmma_f32_16x16x32_bf16.
__global__ __launch_bounds__(128) void moe_attn_kernel(
    const float* __restrict__ x, const int* __restrict__ am,
    const float* __restrict__ Wq, const float* __restrict__ Wk,
    const float* __restrict__ Wv, const float* __restrict__ gate,
    float* __restrict__ out) {
  __shared__ SharedBuf sb;

  const int blk  = blockIdx.x;
  const int h    = blk & (H_ - 1);
  const int span = blk >> 4;
  const int b = span / NSPAN_, s = span % NSPAN_;
  const int tid  = threadIdx.x;
  const int lane = tid & 31;
  const int wv   = tid >> 5;                   // wave id 0..3

  // top-2 gate for this span
  const float* gp = gate + span * 4;
  const int   e0 = ((const int*)gp)[0];
  const int   e1 = ((const int*)gp)[1];
  const float g0 = gp[2], g1 = gp[3];

  const float* xs = x + ((size_t)b * T_ + (size_t)s * SPAN_) * (size_t)D_;
  const size_t wbase = (size_t)h * HD_ * D_;   // head's 64 output rows
  const float* wq0 = Wq + (size_t)e0 * D_ * D_ + wbase;
  const float* wq1 = Wq + (size_t)e1 * D_ * D_ + wbase;
  const float* wk0 = Wk + (size_t)e0 * D_ * D_ + wbase;
  const float* wk1 = Wk + (size_t)e1 * D_ * D_ + wbase;
  const float* wv0 = Wv + (size_t)e0 * D_ * D_ + wbase;
  const float* wv1 = Wv + (size_t)e1 * D_ * D_ + wbase;

  if (tid < SPAN_) sb.maskf[tid] = (float)am[b * T_ + s * SPAN_ + tid];

  v8f accQ[4], accK[4], accV[4];
#pragma unroll
  for (int i = 0; i < 4; ++i) {
    accQ[i] = v8f{0.f, 0.f, 0.f, 0.f, 0.f, 0.f, 0.f, 0.f};
    accK[i] = accQ[i];
    accV[i] = accQ[i];
  }

  // staging map: thread -> (row r 0..63, 16-col group cg)
  const int r  = tid >> 1;
  const int cg = (tid & 1) << 4;

  // ---------------- projection: stream K-dim in blocks of 32 ----------------
  for (int ks = 0; ks < KSTEPS; ++ks) {
    const int k0 = ks * 32;

    {  // x block -> bf16 LDS
      const float* src = xs + (size_t)r * D_ + k0 + cg;
      __bf16* dst = sb.xblk + r * XBS + cg;
#pragma unroll
      for (int i = 0; i < 16; ++i) dst[i] = (__bf16)src[i];
    }
    {  // combined top-2 weight blocks -> bf16 LDS (Q, K, V)
      const size_t off = (size_t)r * D_ + k0 + cg;
      const float *q0 = wq0 + off, *q1 = wq1 + off;
      const float *k0p = wk0 + off, *k1p = wk1 + off;
      const float *v0 = wv0 + off, *v1 = wv1 + off;
      __bf16* dq = sb.wc[0] + r * WCS + cg;
      __bf16* dk = sb.wc[1] + r * WCS + cg;
      __bf16* dv = sb.wc[2] + r * WCS + cg;
#pragma unroll
      for (int i = 0; i < 16; ++i) {
        dq[i] = (__bf16)(g0 * q0[i] + g1 * q1[i]);
        dk[i] = (__bf16)(g0 * k0p[i] + g1 * k1p[i]);
        dv[i] = (__bf16)(g0 * v0[i] + g1 * v1[i]);
      }
      if (ks + 1 < KSTEPS) {  // prefetch next K-block lines into L2/L0
        __builtin_prefetch(xs + (size_t)r * D_ + k0 + 32 + cg, 0, 1);
        __builtin_prefetch(q0 + 32, 0, 1);
        __builtin_prefetch(q1 + 32, 0, 1);
        __builtin_prefetch(k0p + 32, 0, 1);
        __builtin_prefetch(k1p + 32, 0, 1);
        __builtin_prefetch(v0 + 32, 0, 1);
        __builtin_prefetch(v1 + 32, 0, 1);
      }
    }
    __syncthreads();

    const v16bf afrag = load_frag_a(sb.xblk + wv * 16 * XBS, XBS, lane);
#pragma unroll
    for (int nt = 0; nt < 4; ++nt) {
      const v16bf bq = load_frag_b(sb.wc[0] + nt * 16 * WCS, WCS, lane);
      accQ[nt] = wmma_bf16(afrag, bq, accQ[nt]);
      const v16bf bk = load_frag_b(sb.wc[1] + nt * 16 * WCS, WCS, lane);
      accK[nt] = wmma_bf16(afrag, bk, accK[nt]);
      const v16bf bv = load_frag_b(sb.wc[2] + nt * 16 * WCS, WCS, lane);
      accV[nt] = wmma_bf16(afrag, bv, accV[nt]);
    }
    __syncthreads();
  }

  // -------- spill Q, K (row-major) and V (transposed) to LDS as bf16 --------
  const int mrow = (lane >> 4) << 3;  // C-layout: vgpr v holds row v (+8 for hi half)
  const int ncol = lane & 15;
#pragma unroll
  for (int nt = 0; nt < 4; ++nt) {
#pragma unroll
    for (int v = 0; v < 8; ++v) {
      const int row = wv * 16 + mrow + v;
      const int col = nt * 16 + ncol;
      sb.qb[row * QS + col]  = (__bf16)accQ[nt][v];
      sb.kb[row * QS + col]  = (__bf16)accK[nt][v];
      sb.vtb[col * QS + row] = (__bf16)accV[nt][v];  // V^T: [d][m]
    }
  }
  __syncthreads();

  // ---------------- scores S = Q K^T / sqrt(hd), via WMMA ----------------
  v8f accS[4];
#pragma unroll
  for (int i = 0; i < 4; ++i) accS[i] = v8f{0.f, 0.f, 0.f, 0.f, 0.f, 0.f, 0.f, 0.f};
#pragma unroll
  for (int dc = 0; dc < 2; ++dc) {  // head-dim chunks of 32
    const v16bf aq = load_frag_a(sb.qb + wv * 16 * QS + dc * 32, QS, lane);
#pragma unroll
    for (int nt = 0; nt < 4; ++nt) {
      // B[d][m] = K[m][d]: column m comes from K row m, d contiguous
      const v16bf bk = load_frag_b(sb.kb + nt * 16 * QS + dc * 32, QS, lane);
      accS[nt] = wmma_bf16(aq, bk, accS[nt]);
    }
  }

  // scale + key mask
#pragma unroll
  for (int nt = 0; nt < 4; ++nt) {
    const float km = (sb.maskf[nt * 16 + ncol] > 0.f) ? 0.f : -3.0e38f;
#pragma unroll
    for (int v = 0; v < 8; ++v) accS[nt][v] = accS[nt][v] * 0.125f + km;
  }

  // ---------------- softmax over keys (per row, fp32 VALU) ----------------
  // Row M=v(+8) lives in 16 lanes x 4 tiles; xor masks <=8 stay within halves.
#pragma unroll
  for (int v = 0; v < 8; ++v) {
    float rmax = fmaxf(fmaxf(accS[0][v], accS[1][v]), fmaxf(accS[2][v], accS[3][v]));
#pragma unroll
    for (int off = 8; off >= 1; off >>= 1)
      rmax = fmaxf(rmax, __shfl_xor(rmax, off, 32));
    float rsum = 0.f;
#pragma unroll
    for (int nt = 0; nt < 4; ++nt) {
      const float e = __expf(accS[nt][v] - rmax);
      accS[nt][v] = e;
      rsum += e;
    }
#pragma unroll
    for (int off = 8; off >= 1; off >>= 1) rsum += __shfl_xor(rsum, off, 32);
    const float rinv = 1.0f / rsum;
#pragma unroll
    for (int nt = 0; nt < 4; ++nt) accS[nt][v] *= rinv;
  }

  // P -> LDS (per-wave region)
  __bf16* pw = sb.pb + wv * 16 * QS;
#pragma unroll
  for (int nt = 0; nt < 4; ++nt)
#pragma unroll
    for (int v = 0; v < 8; ++v)
      pw[(mrow + v) * QS + nt * 16 + ncol] = (__bf16)accS[nt][v];
  __syncthreads();

  // ---------------- context = P V, via WMMA (V^T layout in LDS) ----------------
  v8f accO[4];
#pragma unroll
  for (int i = 0; i < 4; ++i) accO[i] = v8f{0.f, 0.f, 0.f, 0.f, 0.f, 0.f, 0.f, 0.f};
#pragma unroll
  for (int mc = 0; mc < 2; ++mc) {  // key chunks of 32
    const v16bf ap = load_frag_a(pw + mc * 32, QS, lane);
#pragma unroll
    for (int nt = 0; nt < 4; ++nt) {
      // B[m][d] = V[m][d] = Vt[d][m]: column d from Vt row d, m contiguous
      const v16bf bv = load_frag_b(sb.vtb + nt * 16 * QS + mc * 32, QS, lane);
      accO[nt] = wmma_bf16(ap, bv, accO[nt]);
    }
  }

  // ---------------- write out fp32 ----------------
  float* op = out + ((size_t)b * T_ + (size_t)s * SPAN_) * (size_t)D_ + (size_t)h * HD_;
#pragma unroll
  for (int nt = 0; nt < 4; ++nt)
#pragma unroll
    for (int v = 0; v < 8; ++v) {
      const int row = wv * 16 + mrow + v;
      const int col = nt * 16 + ncol;
      op[(size_t)row * D_ + col] = accO[nt][v];
    }
}

// ================================ launcher ================================
extern "C" void kernel_launch(void* const* d_in, const int* in_sizes, int n_in,
                              void* d_out, int out_size, void* d_ws, size_t ws_size,
                              hipStream_t stream) {
  const float* x   = (const float*)d_in[0];
  const int*   am  = (const int*)d_in[1];
  const float* Wq  = (const float*)d_in[2];
  const float* Wk  = (const float*)d_in[3];
  const float* Wv  = (const float*)d_in[4];
  const float* g1  = (const float*)d_in[5];
  const float* gb1 = (const float*)d_in[6];
  const float* g2  = (const float*)d_in[7];
  const float* gb2 = (const float*)d_in[8];
  float* out  = (float*)d_out;
  float* gate = (float*)d_ws;  // 256 spans * 4 floats = 4 KB

  gate_kernel<<<B_ * NSPAN_, 128, 0, stream>>>(x, am, g1, gb1, g2, gb2, gate);
  moe_attn_kernel<<<B_ * NSPAN_ * H_, 128, 0, stream>>>(x, am, Wq, Wk, Wv, gate, out);
}
